// Fusion_Attention_57234734187137
// MI455X (gfx1250) — matrix operand order repaired and checked by reading, not compiled, not run on hardware
//
#include <hip/hip_runtime.h>

#define B_       4
#define H_       8
#define N_       2048
#define D_       64
// (1/sqrt(64)) * log2(e): scores produced directly in exp2 domain
#define QSCALE   0.18033688011112042f

#define BLOCK_M  128                // 8 waves x 16 query rows
#define BLOCK_N  64
#define NWAVES   8
#define NTHREADS 256
#define KPAD     8
#define KSTRIDE  (D_ + KPAD)        // 72 halfs = 144 B rows (16B aligned)
#define VSTRIDE  (BLOCK_N + KPAD)   // 72
#define PSTRIDE  (BLOCK_N + KPAD)   // 72
#define NB       (N_ / BLOCK_N)     // 32
#define STAGE_IT ((BLOCK_N * D_ / 8) / NTHREADS)   // 2

typedef __attribute__((ext_vector_type(16))) _Float16 v16h;
typedef __attribute__((ext_vector_type(8)))  _Float16 v8h;
typedef __attribute__((ext_vector_type(2)))  _Float16 v2h;
typedef __attribute__((ext_vector_type(8)))  float    v8f;

union F16Frag {
    v16h     v;
    v8h      h8[2];
    _Float16 e[16];
};

__global__ __launch_bounds__(NTHREADS)
void fattn_wmma_gfx1250(const float* __restrict__ q,
                        const float* __restrict__ kg,
                        const float* __restrict__ vg,
                        float* __restrict__ out)
{
    // 55.3 KB static LDS: double-buffered f16 K/Vt tiles + per-wave P scratch
    __shared__ alignas(16) _Float16 Ksh [2][BLOCK_N * KSTRIDE];   // K  [key][d]
    __shared__ alignas(16) _Float16 VTsh[2][D_      * VSTRIDE];   // Vt [d][key]
    __shared__ alignas(16) _Float16 Psh [NWAVES * 16 * PSTRIDE];

    const int tid  = threadIdx.x;
    const int lane = tid & 31;
    const int wave = tid >> 5;
    const int lh   = lane >> 4;   // lane half (0/1)
    const int ln   = lane & 15;

    const size_t base  = (size_t)blockIdx.y * (size_t)(N_ * D_);
    const int    mbase = blockIdx.x * BLOCK_M + wave * 16;

    // ---- Q: two 16x32 f16 A-fragments, pre-scaled into exp2 domain
    F16Frag qa[2];
    {
        const float* qrow = q + base + (size_t)(mbase + ln) * D_;
#pragma unroll
        for (int f = 0; f < 2; ++f) {
            const int dbase = f * 32;
            const float4 a0 = *(const float4*)(qrow + dbase + 8 * lh);
            const float4 a1 = *(const float4*)(qrow + dbase + 8 * lh + 4);
            const float4 a2 = *(const float4*)(qrow + dbase + 16 + 8 * lh);
            const float4 a3 = *(const float4*)(qrow + dbase + 16 + 8 * lh + 4);
            const float t[16] = { a0.x, a0.y, a0.z, a0.w,  a1.x, a1.y, a1.z, a1.w,
                                  a2.x, a2.y, a2.z, a2.w,  a3.x, a3.y, a3.z, a3.w };
#pragma unroll
            for (int j = 0; j < 16; ++j) qa[f].e[j] = (_Float16)(t[j] * QSCALE);
        }
    }

    v8f o[4] = {};
    float mrow[8], lpart[8];     // lpart: per-lane PARTIAL row sums (reduced once at end)
#pragma unroll
    for (int r = 0; r < 8; ++r) { mrow[r] = -__builtin_inff(); lpart[r] = 0.0f; }

    // ---- cooperative stage: f32 -> f16, K row-major (b128), V transposed (paired b32)
    auto stage = [&](int jb, int buf) {
        const float* kblk = kg + base + (size_t)jb * (BLOCK_N * D_);
        const float* vblk = vg + base + (size_t)jb * (BLOCK_N * D_);
        _Float16* kd = &Ksh [buf][0];
        _Float16* vd = &VTsh[buf][0];
#pragma unroll
        for (int i = 0; i < STAGE_IT; ++i) {
            const int c = tid + i * NTHREADS;          // 0..511 chunks
            {   // K: one chunk = 8 contiguous d of one key -> single ds_store_b128
                const int kk = c >> 3;
                const int dd = (c & 7) << 3;
                const float4 a = *(const float4*)(kblk + kk * D_ + dd);
                const float4 b = *(const float4*)(kblk + kk * D_ + dd + 4);
                const v8h p = { (_Float16)a.x, (_Float16)a.y, (_Float16)a.z, (_Float16)a.w,
                                (_Float16)b.x, (_Float16)b.y, (_Float16)b.z, (_Float16)b.w };
                *(v8h*)&kd[kk * KSTRIDE + dd] = p;
            }
            {   // V: one chunk = 2 keys x 4 d -> 4 paired ds_store_b32 (transposed)
                const int kk = (c >> 4) << 1;
                const int dd = (c & 15) << 2;
                const float4 a = *(const float4*)(vblk + kk * D_ + dd);
                const float4 b = *(const float4*)(vblk + (kk + 1) * D_ + dd);
                const v2h p0 = { (_Float16)a.x, (_Float16)b.x };
                const v2h p1 = { (_Float16)a.y, (_Float16)b.y };
                const v2h p2 = { (_Float16)a.z, (_Float16)b.z };
                const v2h p3 = { (_Float16)a.w, (_Float16)b.w };
                *(v2h*)&vd[(dd + 0) * VSTRIDE + kk] = p0;
                *(v2h*)&vd[(dd + 1) * VSTRIDE + kk] = p1;
                *(v2h*)&vd[(dd + 2) * VSTRIDE + kk] = p2;
                *(v2h*)&vd[(dd + 3) * VSTRIDE + kk] = p3;
            }
        }
    };

    stage(0, 0);
    __syncthreads();

    for (int jb = 0; jb < NB; ++jb) {
        const int cur = jb & 1;

        // stage next block into the other buffer (overlaps with this block's WMMA)
        if (jb + 1 < NB) {
            stage(jb + 1, cur ^ 1);
            if (jb + 2 < NB) {   // warm L2 two blocks ahead (global_prefetch_b8)
                __builtin_prefetch(kg + base + (size_t)(jb + 2) * (BLOCK_N * D_) + tid * 16, 0, 1);
                __builtin_prefetch(vg + base + (size_t)(jb + 2) * (BLOCK_N * D_) + tid * 16, 0, 1);
            }
        }

        const _Float16* kd = &Ksh [cur][0];
        const _Float16* vd = &VTsh[cur][0];

        // ---- S = (Q*scale*log2e) . K^T : four 16x16 tiles over 64 keys
        v8f s[4];
#pragma unroll
        for (int nt = 0; nt < 4; ++nt) {
            const int key = nt * 16 + ln;
            F16Frag b0, b1;
            b0.h8[0] = *(const v8h*)&kd[key * KSTRIDE + lh * 8];
            b0.h8[1] = *(const v8h*)&kd[key * KSTRIDE + lh * 8 + 16];
            b1.h8[0] = *(const v8h*)&kd[key * KSTRIDE + 32 + lh * 8];
            b1.h8[1] = *(const v8h*)&kd[key * KSTRIDE + 32 + lh * 8 + 16];
            v8f c = {};
            c = __builtin_amdgcn_wmma_f32_16x16x32_f16(false, qa[0].v, false, b0.v,
                                                       (short)0, c, false, false);
            c = __builtin_amdgcn_wmma_f32_16x16x32_f16(false, qa[1].v, false, b1.v,
                                                       (short)0, c, false, false);
            s[nt] = c;
        }

        // ---- online softmax in exp2 domain; only the MAX needs a cross-lane reduce
        float alpha[8];
#pragma unroll
        for (int r = 0; r < 8; ++r) {
            float mx = fmaxf(fmaxf(s[0][r], s[1][r]), fmaxf(s[2][r], s[3][r]));
#pragma unroll
            for (int off = 1; off < 16; off <<= 1)
                mx = fmaxf(mx, __shfl_xor(mx, off, 16));
            const float mnew = fmaxf(mrow[r], mx);
            alpha[r] = __builtin_amdgcn_exp2f(mrow[r] - mnew);
            mrow[r]  = mnew;
            float ps = 0.0f;
#pragma unroll
            for (int nt = 0; nt < 4; ++nt) {
                const float p = __builtin_amdgcn_exp2f(s[nt][r] - mnew);
                s[nt][r] = p;
                ps += p;
            }
            lpart[r] = lpart[r] * alpha[r] + ps;   // per-lane partial; reduced at end
        }
#pragma unroll
        for (int dt = 0; dt < 4; ++dt)
#pragma unroll
            for (int r = 0; r < 8; ++r)
                o[dt][r] = o[dt][r] * alpha[r];

        // ---- relayout P: C-frag -> LDS -> A-frag (wave-private, dscnt fence only)
        // packed conversion: one v_cvt_pk_f16_f32 per column pair
        _Float16* pw = &Psh[wave * 16 * PSTRIDE];
#pragma unroll
        for (int nt = 0; nt < 4; nt += 2) {
#pragma unroll
            for (int r = 0; r < 8; ++r) {
                const v2h pk = { (_Float16)s[nt][r], (_Float16)s[nt + 1][r] };
                const int rowoff = (r + 8 * lh) * PSTRIDE + ln;
                pw[rowoff + nt * 16]      = pk.x;
                pw[rowoff + nt * 16 + 16] = pk.y;   // ds_store_b16_d16_hi candidate
            }
        }

        asm volatile("s_wait_dscnt 0" ::: "memory");

        // ---- O += P . V : f-outer so only one P A-fragment is live at a time
#pragma unroll
        for (int f = 0; f < 2; ++f) {
            const int kfb = f * 32;
            F16Frag pa;
            pa.h8[0] = *(const v8h*)&pw[ln * PSTRIDE + kfb + 8 * lh];
            pa.h8[1] = *(const v8h*)&pw[ln * PSTRIDE + kfb + 16 + 8 * lh];
#pragma unroll
            for (int dt = 0; dt < 4; ++dt) {
                const int dcol = dt * 16 + ln;
                F16Frag vb;
                vb.h8[0] = *(const v8h*)&vd[dcol * VSTRIDE + kfb + lh * 8];
                vb.h8[1] = *(const v8h*)&vd[dcol * VSTRIDE + kfb + lh * 8 + 16];
                o[dt] = __builtin_amdgcn_wmma_f32_16x16x32_f16(false, pa.v, false, vb.v,
                                                               (short)0, o[dt], false, false);
            }
        }

        __syncthreads();   // single barrier per key block (double-buffered staging)
    }

    // ---- epilogue: reduce deferred row sums once, normalize, store
#pragma unroll
    for (int r = 0; r < 8; ++r) {
        float ls = lpart[r];
#pragma unroll
        for (int off = 1; off < 16; off <<= 1)
            ls += __shfl_xor(ls, off, 16);
        lpart[r] = 1.0f / ls;
    }
#pragma unroll
    for (int dt = 0; dt < 4; ++dt) {
#pragma unroll
        for (int r = 0; r < 8; ++r) {
            const int row = mbase + r + 8 * lh;
            out[base + (size_t)row * D_ + dt * 16 + ln] = o[dt][r] * lpart[r];
        }
    }
}

extern "C" void kernel_launch(void* const* d_in, const int* in_sizes, int n_in,
                              void* d_out, int out_size, void* d_ws, size_t ws_size,
                              hipStream_t stream) {
    (void)in_sizes; (void)n_in; (void)out_size; (void)d_ws; (void)ws_size;
    const float* q = (const float*)d_in[0];
    const float* k = (const float*)d_in[1];
    const float* v = (const float*)d_in[2];
    float* out = (float*)d_out;
    dim3 grid(N_ / BLOCK_M, B_ * H_);
    dim3 block(NTHREADS);
    fattn_wmma_gfx1250<<<grid, block, 0, stream>>>(q, k, v, out);
}
